// DiffusionClassifier_38817914421768
// MI455X (gfx1250) — matrix operand order repaired and checked by reading, not compiled, not run on hardware
//
#include <hip/hip_runtime.h>

// ---------------------------------------------------------------------------
// DiffusionClassifier for MI455X (gfx1250, wave32, WMMA).
// All big GEMMs (conv2/conv3 implicit GEMM, 3-layer per-class MLP) run on
// v_wmma_f32_16x16x32_bf16 with f32 accumulation. 126 GFLOP total, ~200MB
// HBM traffic -> compute-bound on the matrix pipe.
// This revision: the WMMA/DS interleaving is pinned with
// __builtin_amdgcn_sched_group_barrier (the pre-RA scheduler kept sinking the
// B-fragment ds_loads onto their consuming WMMA, serializing the XDL pipe on
// s_wait_dscnt 0x0). Pipeline: 6 DS reads up front, then {1 WMMA, 2 DS reads}
// x6, then 2 WMMAs -> each WMMA's operands were loaded two WMMAs earlier.
// ---------------------------------------------------------------------------

typedef __attribute__((ext_vector_type(16))) __bf16   v16bf;
typedef __attribute__((ext_vector_type(8)))  float    v8f;
typedef __attribute__((ext_vector_type(4)))  unsigned u32x4;

union FragBF { v16bf v; u32x4 q[2]; };

static __device__ __forceinline__ unsigned short f2bf(float f) {
  unsigned int u = __float_as_uint(f);
  unsigned int r = (u + 0x7FFFu + ((u >> 16) & 1u)) >> 16;  // RNE
  return (unsigned short)r;
}
static __device__ __forceinline__ float bf2f(unsigned short s) {
  return __uint_as_float(((unsigned int)s) << 16);
}

static __device__ __forceinline__ float alpha_cumprod(int t) {
  float acp = 1.0f;
  for (int i = 0; i <= t; ++i) {
    float beta = 1e-4f + (0.02f - 1e-4f) * (float)i / 9.0f;  // linspace(1e-4,0.02,10)
    acp *= (1.0f - beta);
  }
  return acp;
}

// ---------------------------------------------------------------------------
// conv1 (3->64, 32x32) + BN + ReLU + 2x2 maxpool, direct. Output bf16 NHWC
// [B,16,16,64]. Lanes share the pixel (c fastest) -> broadcast input reads.
// ---------------------------------------------------------------------------
__global__ __launch_bounds__(256) void conv1_fused(
    const float* __restrict__ x, const float* __restrict__ w,
    const float* __restrict__ bb, const float* __restrict__ g,
    const float* __restrict__ be, const float* __restrict__ mn,
    const float* __restrict__ vr, unsigned short* __restrict__ out) {
  int t  = blockIdx.x * 256 + threadIdx.x;   // 1024*16*16*64 threads
  int c  = t & 63;
  int px = (t >> 6) & 15;
  int py = (t >> 10) & 15;
  int b  = t >> 14;
  float sc = g[c] * rsqrtf(vr[c] + 1e-5f);
  float bi = (bb[c] - mn[c]) * sc + be[c];
  float mx = 0.0f;  // post-ReLU values are >= 0
  for (int oy2 = 0; oy2 < 2; ++oy2)
    for (int ox2 = 0; ox2 < 2; ++ox2) {
      int oy = 2 * py + oy2, ox = 2 * px + ox2;
      float s = 0.0f;
#pragma unroll
      for (int ci = 0; ci < 3; ++ci)
#pragma unroll
        for (int ky = 0; ky < 3; ++ky) {
          int iy = oy + ky - 1;
          if (iy < 0 || iy > 31) continue;
#pragma unroll
          for (int kx = 0; kx < 3; ++kx) {
            int ix = ox + kx - 1;
            if (ix < 0 || ix > 31) continue;
            s += x[((b * 3 + ci) * 32 + iy) * 32 + ix] *
                 w[((c * 3 + ci) * 3 + ky) * 3 + kx];
          }
        }
      mx = fmaxf(mx, fmaxf(s * sc + bi, 0.0f));
    }
  out[t] = f2bf(mx);  // NHWC [b][py][px][c]
}

// ---------------------------------------------------------------------------
// Generic WMMA GEMM.  C = A(MxK) * B(KxN), B supplied transposed [N][K] bf16.
// Block: 256 threads = 8 waves, tile 128(M) x 128(N), K chunk = 32.
// CONV=true: A is implicit-im2col from bf16 NHWC input [B,H,W,Cin],
//            K = 9*Cin ordered (dy,dx,ci) so a 32-chunk is contiguous.
// EPI 0: f32 scale*acc+bias, ReLU, bf16 store (conv+BN).
// EPI 1: acc+bias, ReLU, bf16 store (MLP hidden layers).
// EPI 2: acc+bias (=score), d=score-noise, sum d^2 over N, atomicAdd err[c,b].
// ---------------------------------------------------------------------------
template <bool CONV, int EPI>
__global__ __launch_bounds__(256) void gemm_wmma(
    const unsigned short* __restrict__ A, const unsigned short* __restrict__ Bt,
    unsigned short* __restrict__ C, const float* __restrict__ scale,
    const float* __restrict__ bias, const float* __restrict__ noise,
    float* __restrict__ err, int M, int N, int K, int H, int W, int Cin,
    long sA, long sB, long sBias, long sC) {
  const int cls = blockIdx.z;
  A    += (size_t)cls * sA;
  Bt   += (size_t)cls * sB;
  bias += (size_t)cls * sBias;
  if (EPI != 2) C += (size_t)cls * sC;

  const int m0   = blockIdx.x * 128;
  const int n0   = blockIdx.y * 128;
  const int tid  = threadIdx.x;
  const int lane = tid & 31;
  const int wv   = tid >> 5;
  const int l16  = lane & 15;
  const bool hi  = lane >= 16;
  const int r    = tid >> 1;   // tile row loaded by this thread (0..127)
  const int p    = tid & 1;    // 32-byte half of the 64B row

  // Double-buffered tiles: per buffer, A tile 128x80B then B tile 128x80B.
  __shared__ __align__(16) char lds[2 * 20480];

  // Per-row image coordinates for the implicit-GEMM path (constant over K).
  int ab = 0, ay = 0, ax = 0;
  if (CONV) {
    int m  = m0 + r;
    int hw = H * W;
    ab = m / hw;
    int rm = m - ab * hw;
    ay = rm / W;
    ax = rm - ay * W;
  }

  // LDS byte offsets (within one 20480B buffer).
  const int stA    = r * 80 + p * 32;          // A staging store
  const int stB    = 10240 + stA;              // B staging store
  const int abase  = (wv * 16 + l16) * 80 + (hi ? 32 : 0);          // A frag
  const int bbase0 = 10240 + l16 * 80 + (hi ? 32 : 0);              // B frag ns=0

  v8f acc[8];
#pragma unroll
  for (int ns = 0; ns < 8; ++ns)
#pragma unroll
    for (int e = 0; e < 8; ++e) acc[ns][e] = 0.0f;

  const int nk = K >> 5;

  // ---- stage chunk 0 (tap dy=0,dx=0,coff=0 for conv) ----
  int dy = 0, dx = 0, coff = 0;
  u32x4 ra0, ra1, rb0, rb1;
  {
    const unsigned short* bp = Bt + (size_t)(n0 + r) * K + p * 16;
    rb0 = ((const u32x4*)bp)[0];
    rb1 = ((const u32x4*)bp)[1];
    if (CONV) {
      int iy = ay - 1, ix = ax - 1;
      bool av = ((unsigned)iy < (unsigned)H) && ((unsigned)ix < (unsigned)W);
      if (av) {
        const unsigned short* ap =
            A + (((size_t)ab * H + iy) * W + ix) * (size_t)Cin + p * 16;
        ra0 = ((const u32x4*)ap)[0];
        ra1 = ((const u32x4*)ap)[1];
      } else {
        ra0[0] = ra0[1] = ra0[2] = ra0[3] = 0u;
        ra1 = ra0;
      }
    } else {
      const unsigned short* ap = A + (size_t)(m0 + r) * K + p * 16;
      ra0 = ((const u32x4*)ap)[0];
      ra1 = ((const u32x4*)ap)[1];
    }
  }
  *(u32x4*)(lds + stA)      = ra0;
  *(u32x4*)(lds + stA + 16) = ra1;
  *(u32x4*)(lds + stB)      = rb0;
  *(u32x4*)(lds + stB + 16) = rb1;

  int cur = 0;
  for (int kk = 0; kk < nk; ++kk) {
    __syncthreads();  // buf[cur] fully staged for all waves

    // ---- issue global loads for chunk kk+1 (overlap with WMMA below) ----
    const bool more = (kk + 1) < nk;
    int ndy = dy, ndx = dx, ncoff = coff;
    if (more) {
      const int k1 = (kk + 1) << 5;
      if (CONV) {
        // division-free tap advance: k = (dy*3+dx)*Cin + coff
        ncoff = coff + 32;
        if (ncoff == Cin) {
          ncoff = 0;
          ndx = dx + 1;
          if (ndx == 3) { ndx = 0; ndy = dy + 1; }
        }
        int iy = ay + ndy - 1, ix = ax + ndx - 1;
        bool av = ((unsigned)iy < (unsigned)H) && ((unsigned)ix < (unsigned)W);
        if (av) {
          const unsigned short* ap =
              A + (((size_t)ab * H + iy) * W + ix) * (size_t)Cin + ncoff + p * 16;
          ra0 = ((const u32x4*)ap)[0];
          ra1 = ((const u32x4*)ap)[1];
        } else {
          ra0[0] = ra0[1] = ra0[2] = ra0[3] = 0u;
          ra1 = ra0;
        }
      } else {
        const unsigned short* ap = A + (size_t)(m0 + r) * K + k1 + p * 16;
        ra0 = ((const u32x4*)ap)[0];
        ra1 = ((const u32x4*)ap)[1];
      }
      const unsigned short* bp = Bt + (size_t)(n0 + r) * K + k1 + p * 16;
      rb0 = ((const u32x4*)bp)[0];
      rb1 = ((const u32x4*)bp)[1];
      if (kk + 2 < nk)
        __builtin_prefetch((const void*)(bp + 32), 0, 3);  // global_prefetch_b8
    }

    // ---- compute chunk kk from buf[cur] ----
    const char* cbuf = lds + cur * 20480;
    // A 16x32 bf16 frag: M=lane%16; lanes<16 hold K{0..15}, else {16..31}
    FragBF af;
    af.q[0] = *(const u32x4*)(cbuf + abase);
    af.q[1] = *(const u32x4*)(cbuf + abase + 16);
    // 2-deep pipeline over B fragments (only 2 frags live at a time).
    FragBF bfr[2];
    bfr[0].q[0] = *(const u32x4*)(cbuf + bbase0);
    bfr[0].q[1] = *(const u32x4*)(cbuf + bbase0 + 16);
#pragma unroll
    for (int ns = 0; ns < 8; ++ns) {
      if (ns + 1 < 8) {
        const char* bb = cbuf + bbase0 + (ns + 1) * 1280;  // 16 rows * 80B
        bfr[(ns + 1) & 1].q[0] = *(const u32x4*)(bb);
        bfr[(ns + 1) & 1].q[1] = *(const u32x4*)(bb + 16);
      }
      acc[ns] = __builtin_amdgcn_wmma_f32_16x16x32_bf16(
          false, af.v, false, bfr[ns & 1].v, (short)0, acc[ns], false, false);
    }
    // Pin the schedule so the pre-RA scheduler cannot sink the B-frag loads
    // onto their consuming WMMA (which serializes the XDL pipe on dscnt 0).
    // Region has 18 DS reads (2 A + 16 B) and 8 WMMAs:
    //   DS x6 ; { WMMA x1 ; DS x2 } x6 ; WMMA x2
    __builtin_amdgcn_sched_group_barrier(0x100, 6, 0);   // 6 DS reads
#pragma unroll
    for (int gi = 0; gi < 6; ++gi) {
      __builtin_amdgcn_sched_group_barrier(0x008, 1, 0); // 1 WMMA
      __builtin_amdgcn_sched_group_barrier(0x100, 2, 0); // 2 DS reads
    }
    __builtin_amdgcn_sched_group_barrier(0x008, 2, 0);   // final 2 WMMAs

    // ---- stage chunk kk+1 into the other buffer (no barrier needed: that
    //      buffer was last read before the barrier at the top of this iter) ----
    if (more) {
      char* nb = lds + (cur ^ 1) * 20480;
      *(u32x4*)(nb + stA)      = ra0;
      *(u32x4*)(nb + stA + 16) = ra1;
      *(u32x4*)(nb + stB)      = rb0;
      *(u32x4*)(nb + stB + 16) = rb1;
      dy = ndy; dx = ndx; coff = ncoff;
    }
    cur ^= 1;
  }

  // -------- epilogues (C layout: col n = lane%16, row m = e + (hi?8:0)) ----
  if (EPI == 0 || EPI == 1) {
#pragma unroll
    for (int ns = 0; ns < 8; ++ns) {
      int ng   = n0 + ns * 16 + l16;
      float sc = (EPI == 0) ? scale[ng] : 1.0f;
      float bi = bias[ng];
#pragma unroll
      for (int e = 0; e < 8; ++e) {
        int mg   = m0 + wv * 16 + (hi ? 8 : 0) + e;
        float vv = fmaxf(acc[ns][e] * sc + bi, 0.0f);
        C[(size_t)mg * N + ng] = f2bf(vv);
      }
    }
  } else {  // EPI 2: fused per-class squared error vs noise
    float red[8];
#pragma unroll
    for (int e = 0; e < 8; ++e) red[e] = 0.0f;
#pragma unroll
    for (int ns = 0; ns < 8; ++ns) {
      int ng   = n0 + ns * 16 + l16;
      float bi = bias[ng];
#pragma unroll
      for (int e = 0; e < 8; ++e) {
        int mg  = m0 + wv * 16 + (hi ? 8 : 0) + e;
        float d = (acc[ns][e] + bi) - noise[(size_t)mg * N + ng];
        red[e] += d * d;
      }
    }
#pragma unroll
    for (int e = 0; e < 8; ++e) {  // reduce over the 16 columns (lane halves)
      float s = red[e];
      s += __shfl_xor(s, 1, 32);
      s += __shfl_xor(s, 2, 32);
      s += __shfl_xor(s, 4, 32);
      s += __shfl_xor(s, 8, 32);
      if (l16 == 0) {
        int mg = m0 + wv * 16 + (hi ? 8 : 0) + e;
        atomicAdd(&err[(size_t)cls * M + mg], s);
      }
    }
  }
}

// ---------------------------------------------------------------------------
// Small prep / glue kernels
// ---------------------------------------------------------------------------
__global__ void bn_prep(const float* __restrict__ g, const float* __restrict__ be,
                        const float* __restrict__ m, const float* __restrict__ v,
                        const float* __restrict__ b, float* __restrict__ scale,
                        float* __restrict__ bias) {
  int c = threadIdx.x;
  float sc = g[c] * rsqrtf(v[c] + 1e-5f);
  scale[c] = sc;
  bias[c]  = (b[c] - m[c]) * sc + be[c];
}

// OIHW f32 -> Bt bf16 [O][9*I] with k = (dy*3+dx)*I + i
__global__ __launch_bounds__(256) void reformat_conv_w(
    const float* __restrict__ w, unsigned short* __restrict__ wt, int O, int I) {
  int t = blockIdx.x * 256 + threadIdx.x;  // O*9*I threads
  int nk = 9 * I;
  int o = t / nk, k = t - o * nk;
  int pp = k / I, i = k - pp * I;
  int dy = pp / 3, dx = pp - dy * 3;
  wt[t] = f2bf(w[((o * I + i) * 3 + dy) * 3 + dx]);
}

// [C][K][N] f32 -> [C][N][K] bf16
__global__ __launch_bounds__(256) void transpose_sw(
    const float* __restrict__ in, unsigned short* __restrict__ out, int Kd, int Nd) {
  long t = (long)blockIdx.x * 256 + threadIdx.x;  // C*Kd*Nd threads
  long kn = (long)Kd * Nd;
  int c = (int)(t / kn);
  long rm = t - (long)c * kn;
  int n = (int)(rm / Kd), k = (int)(rm - (long)n * Kd);
  out[t] = f2bf(in[((long)c * Kd + k) * Nd + n]);
}

// 2x2 maxpool on bf16 NHWC: in [B,2H,2W,C] -> out [B,H,W,C]
__global__ __launch_bounds__(256) void pool2x2(
    const unsigned short* __restrict__ in, unsigned short* __restrict__ out,
    int H, int W, int C) {
  long t = (long)blockIdx.x * 256 + threadIdx.x;  // B*H*W*C threads
  int c = (int)(t % C);
  long q = t / C;
  int px = (int)(q % W); q /= W;
  int py = (int)(q % H);
  int b  = (int)(q / H);
  long base = (((long)b * (2 * H) + 2 * py) * (2 * W) + 2 * px) * C + c;
  long rs = (long)2 * W * C;
  float m0 = fmaxf(bf2f(in[base]), bf2f(in[base + C]));
  float m1 = fmaxf(bf2f(in[base + rs]), bf2f(in[base + rs + C]));
  out[t] = f2bf(fmaxf(m0, m1));
}

// pool3 + forward diffusion: pre3 bf16 [B,8,8,256] -> comb[b][f] = bf16(x_t),
// f = c*16 + py*4 + px (reference NCHW reshape order). comb ld = 4224.
__global__ __launch_bounds__(256) void pool3_xt(
    const unsigned short* __restrict__ pre3, const float* __restrict__ noise,
    const int* __restrict__ tarr, unsigned short* __restrict__ comb) {
  long t = (long)blockIdx.x * 256 + threadIdx.x;  // 1024*4096 threads
  int c = (int)(t & 255);
  long q = t >> 8;
  int px = (int)(q & 3);
  int py = (int)((q >> 2) & 3);
  int b  = (int)(q >> 4);
  long base = (((long)b * 8 + 2 * py) * 8 + 2 * px) * 256 + c;
  float m0 = fmaxf(bf2f(pre3[base]), bf2f(pre3[base + 256]));
  float m1 = fmaxf(bf2f(pre3[base + 2048]), bf2f(pre3[base + 2048 + 256]));
  float feat = fmaxf(m0, m1);
  int f = c * 16 + py * 4 + px;
  float a  = alpha_cumprod(tarr[b]);
  float xt = sqrtf(a) * feat + sqrtf(1.0f - a) * noise[(long)b * 4096 + f];
  comb[(long)b * 4224 + f] = f2bf(xt);
}

// sinusoidal time embedding into comb[:, 4096:4224]
__global__ __launch_bounds__(256) void temb_k(const int* __restrict__ tarr,
                                              unsigned short* __restrict__ comb) {
  int t = blockIdx.x * 256 + threadIdx.x;  // 1024*128 threads
  int b = t >> 7, j = t & 127;
  float tf = (float)tarr[b];
  float v;
  if (j < 64) {
    v = sinf(tf * expf(-logf(10000.0f) * (float)j / 63.0f));
  } else {
    v = cosf(tf * expf(-logf(10000.0f) * (float)(j - 64) / 63.0f));
  }
  comb[(long)b * 4224 + 4096 + j] = f2bf(v);
}

// logits[b][c] = -((1-a)/a) * sum_f (noise - score)^2   (== -sum (x0-feat)^2)
__global__ __launch_bounds__(256) void logits_k(const float* __restrict__ err,
                                                const int* __restrict__ tarr,
                                                float* __restrict__ out) {
  int t = blockIdx.x * 256 + threadIdx.x;  // 10240 threads
  int c = t % 10, b = t / 10;
  float a = alpha_cumprod(tarr[b]);
  out[t] = -((1.0f - a) / a) * err[(long)c * 1024 + b];
}

// ---------------------------------------------------------------------------
// Host orchestration
// ---------------------------------------------------------------------------
extern "C" void kernel_launch(void* const* d_in, const int* in_sizes, int n_in,
                              void* d_out, int out_size, void* d_ws,
                              size_t ws_size, hipStream_t stream) {
  (void)in_sizes; (void)n_in; (void)out_size;
  const float* x     = (const float*)d_in[0];
  const float* noise = (const float*)d_in[1];
  const int*   tarr  = (const int*)d_in[2];
  const float *w1 = (const float*)d_in[3],  *b1 = (const float*)d_in[4],
              *g1 = (const float*)d_in[5],  *be1 = (const float*)d_in[6],
              *m1 = (const float*)d_in[7],  *v1 = (const float*)d_in[8];
  const float *w2 = (const float*)d_in[9],  *b2 = (const float*)d_in[10],
              *g2 = (const float*)d_in[11], *be2 = (const float*)d_in[12],
              *m2 = (const float*)d_in[13], *v2 = (const float*)d_in[14];
  const float *w3 = (const float*)d_in[15], *b3 = (const float*)d_in[16],
              *g3 = (const float*)d_in[17], *be3 = (const float*)d_in[18],
              *m3 = (const float*)d_in[19], *v3 = (const float*)d_in[20];
  const float *sW1 = (const float*)d_in[21], *sb1 = (const float*)d_in[22],
              *sW2 = (const float*)d_in[23], *sb2 = (const float*)d_in[24],
              *sW3 = (const float*)d_in[25], *sb3 = (const float*)d_in[26];

  char* base = (char*)d_ws;
  size_t off = 0;
  auto alloc = [&](size_t bytes) -> void* {
    void* p = base + off;
    off += (bytes + 255) & ~(size_t)255;
    return p;
  };
  unsigned short* out1 = (unsigned short*)alloc((size_t)1024 * 16 * 16 * 64 * 2);
  unsigned short* w2t  = (unsigned short*)alloc((size_t)128 * 576 * 2);
  float* sc2 = (float*)alloc(128 * 4);
  float* bi2 = (float*)alloc(128 * 4);
  unsigned short* pre2 = (unsigned short*)alloc((size_t)1024 * 16 * 16 * 128 * 2);
  unsigned short* out2 = (unsigned short*)alloc((size_t)1024 * 8 * 8 * 128 * 2);
  unsigned short* w3t  = (unsigned short*)alloc((size_t)256 * 1152 * 2);
  float* sc3 = (float*)alloc(256 * 4);
  float* bi3 = (float*)alloc(256 * 4);
  unsigned short* pre3 = (unsigned short*)alloc((size_t)1024 * 8 * 8 * 256 * 2);
  unsigned short* comb = (unsigned short*)alloc((size_t)1024 * 4224 * 2);
  unsigned short* sW1t = (unsigned short*)alloc((size_t)10 * 256 * 4224 * 2);
  unsigned short* sW2t = (unsigned short*)alloc((size_t)10 * 256 * 256 * 2);
  unsigned short* sW3t = (unsigned short*)alloc((size_t)10 * 4096 * 256 * 2);
  unsigned short* h1   = (unsigned short*)alloc((size_t)10 * 1024 * 256 * 2);
  unsigned short* h2   = (unsigned short*)alloc((size_t)10 * 1024 * 256 * 2);
  float* err = (float*)alloc((size_t)10 * 1024 * 4);
  if (off > ws_size) return;  // ~215 MB needed

  hipMemsetAsync(err, 0, (size_t)10 * 1024 * 4, stream);

  bn_prep<<<1, 128, 0, stream>>>(g2, be2, m2, v2, b2, sc2, bi2);
  bn_prep<<<1, 256, 0, stream>>>(g3, be3, m3, v3, b3, sc3, bi3);
  reformat_conv_w<<<(128 * 9 * 64) / 256, 256, 0, stream>>>(w2, w2t, 128, 64);
  reformat_conv_w<<<(256 * 9 * 128) / 256, 256, 0, stream>>>(w3, w3t, 256, 128);
  transpose_sw<<<(10L * 4224 * 256) / 256, 256, 0, stream>>>(sW1, sW1t, 4224, 256);
  transpose_sw<<<(10L * 256 * 256) / 256, 256, 0, stream>>>(sW2, sW2t, 256, 256);
  transpose_sw<<<(10L * 256 * 4096) / 256, 256, 0, stream>>>(sW3, sW3t, 256, 4096);

  // conv backbone
  conv1_fused<<<65536, 256, 0, stream>>>(x, w1, b1, g1, be1, m1, v1, out1);
  gemm_wmma<true, 0><<<dim3(2048, 1, 1), 256, 0, stream>>>(
      out1, w2t, pre2, sc2, bi2, nullptr, nullptr,
      1024 * 256, 128, 576, 16, 16, 64, 0, 0, 0, 0);
  pool2x2<<<(1024L * 8 * 8 * 128) / 256, 256, 0, stream>>>(pre2, out2, 8, 8, 128);
  gemm_wmma<true, 0><<<dim3(512, 2, 1), 256, 0, stream>>>(
      out2, w3t, pre3, sc3, bi3, nullptr, nullptr,
      1024 * 64, 256, 1152, 8, 8, 128, 0, 0, 0, 0);

  // features -> x_t -> comb
  pool3_xt<<<(1024L * 4096) / 256, 256, 0, stream>>>(pre3, noise, tarr, comb);
  temb_k<<<(1024 * 128) / 256, 256, 0, stream>>>(tarr, comb);

  // per-class score MLP (grid.z = 10 classes)
  gemm_wmma<false, 1><<<dim3(8, 2, 10), 256, 0, stream>>>(
      comb, sW1t, h1, nullptr, sb1, nullptr, nullptr,
      1024, 256, 4224, 0, 0, 0, 0, (long)256 * 4224, 256, (long)1024 * 256);
  gemm_wmma<false, 1><<<dim3(8, 2, 10), 256, 0, stream>>>(
      h1, sW2t, h2, nullptr, sb2, nullptr, nullptr,
      1024, 256, 256, 0, 0, 0, (long)1024 * 256, (long)256 * 256, 256,
      (long)1024 * 256);
  gemm_wmma<false, 2><<<dim3(8, 32, 10), 256, 0, stream>>>(
      h2, sW3t, nullptr, nullptr, sb3, noise, err,
      1024, 4096, 256, 0, 0, 0, (long)1024 * 256, (long)4096 * 256, 4096, 0);

  logits_k<<<40, 256, 0, stream>>>(err, tarr, (float*)d_out);
}
